// VectorQuantizer_57655640981566
// MI455X (gfx1250) — matrix operand order repaired and checked by reading, not compiled, not run on hardware
//
#include <hip/hip_runtime.h>
#include <hip/hip_bf16.h>
#include <math.h>

typedef __attribute__((ext_vector_type(16))) __bf16 v16bf;
typedef __attribute__((ext_vector_type(8)))  float  v8f;

#define B_   8
#define D_   256
#define L_   4096
#define K_   4096
#define N_   (B_ * L_)        // 32768 rows
#define M_   (B_ * D_ * L_)   // 8388608 output elements
#define KT_  (K_ / 16)        // 256 column tiles
#define NPART 8192

// workspace byte offsets (all 32B aligned)
#define OFF_WNINV 0u
#define OFF_XNINV 16384u
#define OFF_WP    147456u     // packed normalized weight, interleaved hi/lo bf16 (4 MB + 16KB prefetch pad)
#define OFF_IDX   4358144u
#define OFF_PART  4489216u
#define OFF_CNT   4521984u

// ---- K1: codebook row inverse L2 norms (one wave per row) ----
__global__ void vq_wnorm(const float* __restrict__ w, float* __restrict__ wninv) {
    int wave = (blockIdx.x * blockDim.x + threadIdx.x) >> 5;   // 0..K_-1
    int lane = threadIdx.x & 31;
    const float* row = w + (size_t)wave * D_;
    float s = 0.f;
    #pragma unroll
    for (int i = 0; i < D_ / 32; ++i) { float v = row[lane + i * 32]; s += v * v; }
    #pragma unroll
    for (int m = 16; m >= 1; m >>= 1) s += __shfl_xor(s, m, 32);
    if (lane == 0) wninv[wave] = 1.f / fmaxf(sqrtf(s), 1e-12f);
}

// ---- K2: per-(b,l) inverse L2 norms of inputs (coalesced over l) ----
__global__ void vq_xnorm(const float* __restrict__ x, float* __restrict__ xninv) {
    int n = blockIdx.x * blockDim.x + threadIdx.x;
    int b = n >> 12, l = n & (L_ - 1);
    const float* p = x + (size_t)b * D_ * L_ + l;
    float s = 0.f;
    for (int d = 0; d < D_; ++d) { float v = p[(size_t)d * L_]; s += v * v; }
    xninv[n] = 1.f / fmaxf(sqrtf(s), 1e-12f);
}

// ---- K3: pack normalized codebook into WMMA B-fragment layout, hi/lo interleaved ----
// B 32x16 bf16 order used here (contiguous k, same as A): lane half = lane>>4, col c = lane&15; per lane 16 consecutive
// k starting at half*16.  hi frag of (kt,step) at elements ((kt*8+step)*64 + lane)*16,
// lo frag at +512 elements.  element d = 32*step + half*16 + e.
__global__ void vq_wpack(const float* __restrict__ w, const float* __restrict__ wninv,
                         __bf16* __restrict__ wp) {
    int tid  = blockIdx.x * blockDim.x + threadIdx.x;   // 0..KT_*8*32-1
    int lane = tid & 31;
    int step = (tid >> 5) & 7;
    int kt   = tid >> 8;
    int half = lane >> 4, c = lane & 15;
    int code = kt * 16 + c;
    float invn = wninv[code];
    const float* row = w + (size_t)code * D_ + 32 * step + half * 16;
    __bf16* oh = wp + ((size_t)(tid >> 5) * 64 + lane) * 16;
    __bf16* ol = oh + 512;
    #pragma unroll
    for (int e = 0; e < 16; ++e) {
        float v  = row[e] * invn;
        __bf16 h = (__bf16)v;
        oh[e] = h;
        ol[e] = (__bf16)(v - (float)h);
    }
}

__global__ void vq_zero(int* __restrict__ cnt) {
    cnt[blockIdx.x * blockDim.x + threadIdx.x] = 0;
}

// ---- K4: fused GEMM (bf16 hi/lo x3 WMMA) + per-row argmax. 4 waves/WG, wave = 16 rows ----
__global__ void __launch_bounds__(128) vq_gemm_argmax(
    const float* __restrict__ x, const float* __restrict__ xninv,
    const __bf16* __restrict__ wp, int* __restrict__ idx) {
    __shared__ v16bf lal[4][8 * 32];     // A-lo fragments, per wave / step / lane
    const int t    = threadIdx.x & 31;
    const int wv   = threadIdx.x >> 5;
    const int tile = blockIdx.x * 4 + wv;
    const int half = t >> 4, r = t & 15;
    const int n0 = tile * 16;
    const int b  = n0 >> 12;          // n0 / L_
    const int l0 = n0 & (L_ - 1);
    const float invn = xninv[n0 + r];
    const float* xb = x + (size_t)b * D_ * L_ + l0 + r;

    // Build A fragments (16x32 bf16 layout): lane row = lane&15 (both halves),
// element e: d = 32*step + half*16 + e.
    v16bf Ah[8];
    #pragma unroll
    for (int s = 0; s < 8; ++s) {
        v16bf ah, al;
        #pragma unroll
        for (int e = 0; e < 16; ++e) {
            int d = 32 * s + half * 16 + e;
            float v  = xb[(size_t)d * L_] * invn;
            __bf16 h = (__bf16)v;
            ah[e] = h;
            al[e] = (__bf16)(v - (float)h);
        }
        Ah[s] = ah;
        lal[wv][s * 32 + t] = al;
    }
    __syncthreads();

    float best[8]; int bidx[8];
    #pragma unroll
    for (int s = 0; s < 8; ++s) { best[s] = -3.0e38f; bidx[s] = 0; }
    const int c = t & 15;

    const __bf16* kb = wp + (size_t)t * 16;   // lane base into packed B
    const __bf16* nx = kb + 1024;             // next-fragment prefetch pointer
    v16bf bhA = *(const v16bf*)(kb);
    v16bf blA = *(const v16bf*)(kb + 512);

    for (int kt = 0; kt < KT_; ++kt) {
        v8f a0 = {0.f,0.f,0.f,0.f,0.f,0.f,0.f,0.f};
        v8f a1 = {0.f,0.f,0.f,0.f,0.f,0.f,0.f,0.f};
        v8f a2 = {0.f,0.f,0.f,0.f,0.f,0.f,0.f,0.f};
        #pragma unroll
        for (int s = 0; s < 8; ++s) {
            v16bf bh = bhA, bl = blA;
            // prefetch next step's fragments (16KB pad past array covers the tail)
            bhA = *(const v16bf*)(nx + (size_t)s * 1024);
            blA = *(const v16bf*)(nx + (size_t)s * 1024 + 512);
            v16bf al = lal[wv][s * 32 + t];
            a0 = __builtin_amdgcn_wmma_f32_16x16x32_bf16(false, Ah[s], false, bh, (short)0, a0, false, false);
            a1 = __builtin_amdgcn_wmma_f32_16x16x32_bf16(false, Ah[s], false, bl, (short)0, a1, false, false);
            a2 = __builtin_amdgcn_wmma_f32_16x16x32_bf16(false, al,    false, bh, (short)0, a2, false, false);
        }
        nx += 8192;
        int cidx = kt * 16 + c;
        #pragma unroll
        for (int s = 0; s < 8; ++s) {
            float v = a0[s] + a1[s] + a2[s];
            if (v > best[s]) { best[s] = v; bidx[s] = cidx; }
        }
    }

    // reduce across the 16 lanes of each half (C layout: col = lane&15, row = s + 8*half)
    #pragma unroll
    for (int m = 8; m >= 1; m >>= 1) {
        #pragma unroll
        for (int s = 0; s < 8; ++s) {
            float ov = __shfl_xor(best[s], m, 32);
            int   oi = __shfl_xor(bidx[s], m, 32);
            if (ov > best[s] || (ov == best[s] && oi < bidx[s])) { best[s] = ov; bidx[s] = oi; }
        }
    }
    if (c == 0) {
        #pragma unroll
        for (int s = 0; s < 8; ++s) idx[n0 + half * 8 + s] = bidx[s];
    }
}

// ---- K5: histogram (int atomics: deterministic) ----
__global__ void vq_hist(const int* __restrict__ idx, int* __restrict__ cnt) {
    int n = blockIdx.x * blockDim.x + threadIdx.x;
    atomicAdd(&cnt[idx[n]], 1);
}

// ---- K6: gather -> transposed output, fused fixed-order MSE partials ----
__global__ void vq_out_mse(const float* __restrict__ x, const float* __restrict__ w,
                           const int* __restrict__ idx, float* __restrict__ out,
                           float* __restrict__ part) {
    __shared__ float sm[256];
    int t = threadIdx.x;
    size_t base = (size_t)blockIdx.x * 256 * 4;
    float acc = 0.f;
    #pragma unroll
    for (int i = 0; i < 4; ++i) {
        size_t tid = base + (size_t)i * 256 + t;   // tid = (b*D + d)*L + l
        int l  = (int)(tid & (L_ - 1));
        int bd = (int)(tid >> 12);
        int d  = bd & (D_ - 1);
        int b  = bd >> 8;
        int n  = b * L_ + l;
        float q = w[(size_t)idx[n] * D_ + d];
        out[tid] = q;                              // q_st value == quantized
        float diff = q - x[tid];
        acc += diff * diff;
    }
    sm[t] = acc;
    __syncthreads();
    for (int off = 128; off >= 1; off >>= 1) {
        if (t < off) sm[t] += sm[t + off];
        __syncthreads();
    }
    if (t == 0) part[blockIdx.x] = sm[0];
}

// ---- K7: finalize loss (1.25 * MSE) and perplexity, fixed-order reductions ----
__global__ void vq_final(const float* __restrict__ part, const int* __restrict__ cnt,
                         float* __restrict__ out) {
    __shared__ float sm[256];
    int t = threadIdx.x;
    float s = 0.f;
    for (int i = t; i < NPART; i += 256) s += part[i];
    sm[t] = s; __syncthreads();
    for (int off = 128; off >= 1; off >>= 1) { if (t < off) sm[t] += sm[t + off]; __syncthreads(); }
    float loss = 1.25f * sm[0] / (float)M_;
    __syncthreads();
    float e = 0.f;
    for (int i = t; i < K_; i += 256) {
        float p = (float)cnt[i] / (float)N_;
        e += p * logf(p + 1e-10f);
    }
    sm[t] = e; __syncthreads();
    for (int off = 128; off >= 1; off >>= 1) { if (t < off) sm[t] += sm[t + off]; __syncthreads(); }
    if (t == 0) { out[M_] = loss; out[M_ + 1] = expf(-sm[0]); }
}

extern "C" void kernel_launch(void* const* d_in, const int* in_sizes, int n_in,
                              void* d_out, int out_size, void* d_ws, size_t ws_size,
                              hipStream_t stream) {
    (void)in_sizes; (void)n_in; (void)out_size; (void)ws_size;
    const float* x = (const float*)d_in[0];     // [B, D, L] fp32
    const float* w = (const float*)d_in[1];     // [K, D]    fp32
    float* out = (float*)d_out;
    char*  ws  = (char*)d_ws;
    float*  wninv = (float*)(ws + OFF_WNINV);
    float*  xninv = (float*)(ws + OFF_XNINV);
    __bf16* wp    = (__bf16*)(ws + OFF_WP);
    int*    idx   = (int*)(ws + OFF_IDX);
    float*  part  = (float*)(ws + OFF_PART);
    int*    cnt   = (int*)(ws + OFF_CNT);

    vq_wnorm      <<<K_ / 8,         256, 0, stream>>>(w, wninv);
    vq_xnorm      <<<N_ / 256,       256, 0, stream>>>(x, xninv);
    vq_wpack      <<<(KT_*8*32)/256, 256, 0, stream>>>(w, wninv, wp);
    vq_zero       <<<K_ / 256,       256, 0, stream>>>(cnt);
    vq_gemm_argmax<<<N_ / 16 / 4,    128, 0, stream>>>(x, xninv, wp, idx);
    vq_hist       <<<N_ / 256,       256, 0, stream>>>(idx, cnt);
    vq_out_mse    <<<NPART,          256, 0, stream>>>(x, w, idx, out, part);
    vq_final      <<<1,              256, 0, stream>>>(part, cnt, out);
}
